// CIN_26568667693444
// MI455X (gfx1250) — matrix operand order repaired and checked by reading, not compiled, not run on hardware
//
#include <hip/hip_runtime.h>

// CIN (xDeepFM) fused two-layer kernel for gfx1250 (MI455X, wave32, WMMA).
// B=2048, F0=39, D=16, H1=H2=128.
//
// Layer k is GEMM: out[h, (b,d)] = W[h, p] * Z[p, (b,d)], with Z an outer
// product built on the fly in LDS per batch. f16 WMMA 16x16x32, f32 accum.

typedef _Float16 v16h __attribute__((ext_vector_type(16)));
typedef _Float16 v8h  __attribute__((ext_vector_type(8)));
typedef float    v8f  __attribute__((ext_vector_type(8)));

#define F0   39
#define DDIM 16
#define HOUT 128
#define K1   1521            // 39*39
#define KP1  1536            // padded to /32
#define K2   4992            // 39*128 (already /32)
#define ZK   1280            // LDS Z-chunk width (multiple of 32)

// ---------- prologue: convert weights f32 -> f16 into workspace ----------
__global__ __launch_bounds__(256) void cin_convert_weights(
    const float* __restrict__ W0, const float* __restrict__ W1,
    _Float16* __restrict__ W0h, _Float16* __restrict__ W1h)
{
    const int N0 = HOUT * KP1;          // 196608 (padded layout [128][1536])
    const int N1 = HOUT * K2;           // 638976 (layout [128][4992])
    int i = blockIdx.x * 256 + threadIdx.x;
    if (i < N0) {
        int h = i / KP1;
        int k = i - h * KP1;
        W0h[i] = (k < K1) ? (_Float16)W0[h * K1 + k] : (_Float16)0.0f;
    } else {
        int j = i - N0;
        if (j < N1) W1h[j] = (_Float16)W1[j];
    }
}

// ---------- main kernel: one workgroup per batch ----------
__global__ __launch_bounds__(256) void cin_kernel(
    const float* __restrict__ x,        // [2048][39][16] f32
    const _Float16* __restrict__ W0h,   // [128][1536] f16 (zero padded)
    const _Float16* __restrict__ W1h,   // [128][4992] f16
    float* __restrict__ out)            // [2048][256] f32
{
    __shared__ _Float16 xh[F0 * DDIM + 16];                 // x_b as f16
    __shared__ _Float16 h1h[HOUT * DDIM];                   // layer-1 result
    __shared__ _Float16 h2h[HOUT * DDIM];                   // layer-2 result
    __shared__ __attribute__((aligned(32))) _Float16 Zt[DDIM * ZK]; // Z^T chunk

    const int b    = blockIdx.x;
    const int tid  = threadIdx.x;
    const int lane = tid & 31;
    const int wv   = tid >> 5;      // 0..7 : M-tile (16 output channels each)
    const int hi   = lane >> 4;     // half-wave select
    const int lrow = lane & 15;     // row within M-tile / column N=d

    // ---- stage x_b into LDS as f16 (contiguous [39][16]) ----
    const float* xb = x + b * (F0 * DDIM);
    for (int i = tid; i < F0 * DDIM; i += 256)
        xh[i] = (_Float16)xb[i];
    __syncthreads();

    // =========================== layer 1 ===========================
    v8f c1 = {0.f, 0.f, 0.f, 0.f, 0.f, 0.f, 0.f, 0.f};
    {
        const _Float16* wrow = W0h + (wv * 16 + lrow) * KP1;
        for (int kbase = 0; kbase < KP1; kbase += ZK) {
            const int kw = (KP1 - kbase < ZK) ? (KP1 - kbase) : ZK;
            // build Zt[d][j] = x[f][d]*x[q][d], p = kbase+j = f*39+q
            for (int d = 0; d < DDIM; ++d) {
                _Float16* zr = &Zt[d * ZK];
                for (int j = tid; j < kw; j += 256) {
                    int p = kbase + j;
                    _Float16 v = (_Float16)0.0f;
                    if (p < K1) {
                        int f = p / F0;
                        int q = p - f * F0;
                        v = xh[f * DDIM + d] * xh[q * DDIM + d];
                    }
                    zr[j] = v;
                }
            }
            __syncthreads();
            const int ntiles = kw >> 5;
            for (int j = 0; j < ntiles; ++j) {
                const v8h* pa = (const v8h*)(wrow + kbase + j * 32 + hi * 16);
                v8h alo = pa[0];
                v8h ahi = pa[1];                      // +8 halves
                v16h a = __builtin_shufflevector(alo, ahi,
                         0,1,2,3,4,5,6,7,8,9,10,11,12,13,14,15);
                v16h bf = *(const v16h*)(&Zt[lrow * ZK + j * 32 + hi * 16]);
                c1 = __builtin_amdgcn_wmma_f32_16x16x32_f16(
                         false, a, false, bf, (short)0, c1, false, false);
            }
            __syncthreads();
        }
    }
    // scatter h1 tile to LDS as f16 (C layout: lane L, vgpr r -> M=r+8*hi, N=lrow)
    for (int r = 0; r < 8; ++r) {
        int h = wv * 16 + r + hi * 8;
        h1h[h * DDIM + lrow] = (_Float16)c1[r];
    }
    __syncthreads();

    // =========================== layer 2 ===========================
    v8f c2 = {0.f, 0.f, 0.f, 0.f, 0.f, 0.f, 0.f, 0.f};
    {
        const _Float16* wrow = W1h + (wv * 16 + lrow) * K2;
        for (int kbase = 0; kbase < K2; kbase += ZK) {
            const int kw = (K2 - kbase < ZK) ? (K2 - kbase) : ZK;
            // Zt[d][j] = x[f][d]*h1[q][d], p = kbase+j = f*128+q
            for (int d = 0; d < DDIM; ++d) {
                _Float16* zr = &Zt[d * ZK];
                for (int j = tid; j < kw; j += 256) {
                    int p = kbase + j;
                    int f = p >> 7;
                    int q = p & 127;
                    zr[j] = xh[f * DDIM + d] * h1h[q * DDIM + d];
                }
            }
            __syncthreads();
            const int ntiles = kw >> 5;
            for (int j = 0; j < ntiles; ++j) {
                const v8h* pa = (const v8h*)(wrow + kbase + j * 32 + hi * 16);
                v8h alo = pa[0];
                v8h ahi = pa[1];
                v16h a = __builtin_shufflevector(alo, ahi,
                         0,1,2,3,4,5,6,7,8,9,10,11,12,13,14,15);
                v16h bf = *(const v16h*)(&Zt[lrow * ZK + j * 32 + hi * 16]);
                c2 = __builtin_amdgcn_wmma_f32_16x16x32_f16(
                         false, a, false, bf, (short)0, c2, false, false);
            }
            __syncthreads();
        }
    }
    for (int r = 0; r < 8; ++r) {
        int h = wv * 16 + r + hi * 8;
        h2h[h * DDIM + lrow] = (_Float16)c2[r];
    }
    __syncthreads();

    // ---- final: out[b][h] = sum_d h(layer)[h][d], concat(h1,h2) ----
    {
        const _Float16* src = (tid < HOUT) ? &h1h[tid * DDIM]
                                           : &h2h[(tid - HOUT) * DDIM];
        float s = 0.f;
        #pragma unroll
        for (int d = 0; d < DDIM; ++d) s += (float)src[d];
        out[b * (2 * HOUT) + tid] = s;
    }
}

extern "C" void kernel_launch(void* const* d_in, const int* in_sizes, int n_in,
                              void* d_out, int out_size, void* d_ws, size_t ws_size,
                              hipStream_t stream) {
    const float* x  = (const float*)d_in[0];   // [2048,39,16]
    const float* W0 = (const float*)d_in[1];   // [128,1521,1]
    const float* W1 = (const float*)d_in[2];   // [128,4992,1]
    float* out = (float*)d_out;                // [2048,256]

    _Float16* W0h = (_Float16*)d_ws;
    _Float16* W1h = (_Float16*)((char*)d_ws + (size_t)HOUT * KP1 * sizeof(_Float16));

    const int convN = HOUT * KP1 + HOUT * K2;  // 835584
    cin_convert_weights<<<(convN + 255) / 256, 256, 0, stream>>>(W0, W1, W0h, W1h);
    cin_kernel<<<2048, 256, 0, stream>>>(x, W0h, W1h, out);
}
